// Correlation_flownet_75548474737360
// MI455X (gfx1250) — matrix operand order repaired and checked by reading, not compiled, not run on hardware
//
#include <hip/hip_runtime.h>
#include <stdint.h>

typedef __bf16 bf16_t;
typedef __attribute__((ext_vector_type(8)))  bf16_t v8bf;
typedef __attribute__((ext_vector_type(16))) bf16_t v16bf;
typedef __attribute__((ext_vector_type(8)))  float  v8f;
typedef __attribute__((ext_vector_type(4)))  unsigned int v4u;
typedef __attribute__((ext_vector_type(4)))  int v4i;
typedef __attribute__((ext_vector_type(8)))  int v8i;

#define MD   40
#define KD   81            // kernel size (2*MD+1)
#define C_   128
#define H_   64
#define W_   96
#define B_   2
#define XP   176           // padded x2 width (96 + 2*MD)
#define BSTR 136           // LDS B/A row stride in bf16 elems (272B = 256B + 16B TDM pad)
#define GSTR 20            // Gband row stride in floats (80B, 16B aligned)

#define BBUF_BYTES (96 * BSTR * 2)                   // 26112 B per B buffer
#define BBUF_ELEMS (96 * BSTR)                       // bf16 elems per B buffer
#define ALDS_OFF   (2 * BBUF_BYTES)                  // 52224
#define GBAND_OFF  (ALDS_OFF + 32 * BSTR * 2)        // 60928 (16B aligned)
#define GDUMP      (2 * KD * GSTR)                   // dump slot (float index in Gband)
#define LDS_TOTAL  (GBAND_OFF + (GDUMP + 16) * 4)    // 73952 B

// -------------------------------------------------------------------------
// Pre-pass: x2 (b,c,h,w) f32  ->  x2t (b, y, xp, c) bf16, xp zero-padded.
// Writes perfectly coalesced; strided reads are L2-resident (x2 = 6.3MB).
// -------------------------------------------------------------------------
__global__ void corr_prepass(const float* __restrict__ x2, bf16_t* __restrict__ x2t) {
  int g  = blockIdx.x * 256 + threadIdx.x;      // 2*64*176*128 = 2,883,584 total
  int c  = g & 127;
  int xp = (g >> 7) % XP;
  int y  = ((g >> 7) / XP) % H_;
  int b  = (g >> 7) / (XP * H_);
  int x  = xp - MD;
  float v = 0.0f;
  if (x >= 0 && x < W_) v = x2[(((size_t)b * C_ + c) * H_ + y) * W_ + x];
  x2t[g] = (bf16_t)v;
}

// Issue one TDM transfer: 96 rows x 256B (bf16) global -> LDS, +16B pad/row.
__device__ __forceinline__ void tdm_load_bslice(uint64_t ga, unsigned lds_off) {
  v4u g0;
  g0[0] = 1u;                                               // count=1, no gather
  g0[1] = lds_off;                                          // LDS dest offset
  g0[2] = (unsigned)ga;                                     // global_addr[31:0]
  g0[3] = (unsigned)((ga >> 32) & 0x1FFFFFFu) | (2u << 30); // addr[56:32] | type=2
  v8i g1;
  g1[0] = (int)((1u << 16)       // data_size = 2B
              | (1u << 20)       // pad_enable
              | (5u << 22)       // pad_interval: every 64 dwords (256B)
              | (3u << 25));     // pad_amount: 4 dwords (16B) -> 272B LDS stride
  g1[1] = (int)(((unsigned)C_ & 0xFFFFu) << 16);            // tensor_dim0 lo16 = 128
  g1[2] = (int)(((unsigned)96u & 0xFFFFu) << 16);           // dim0 hi | tensor_dim1 lo16 = 96
  g1[3] = (int)((unsigned)C_ << 16);                        // dim1 hi | tile_dim0 = 128
  g1[4] = 96;                                               // tile_dim1 = 96, tile_dim2 = 0
  g1[5] = C_;                                               // tensor_dim0_stride = 128 elems
  g1[6] = 0;
  g1[7] = 0;
  v4i zz4 = {0, 0, 0, 0};
  v8i zz8 = {0, 0, 0, 0, 0, 0, 0, 0};
  __builtin_amdgcn_tensor_load_to_lds(g0, g1, zz4, zz4, zz8, 0);
}

// -------------------------------------------------------------------------
// Main kernel: per WG = (b, h-pair, 16-wide w tile).
// Double-buffered TDM feeds the 96x128 bf16 B slice; 6 waves x (2 M-tiles x
// 4 k-chunks) of v_wmma_f32_16x16x32_bf16 compute G = A(32x128)*B(128x96);
// the 81-wide diagonal band is staged in LDS and stored as float4 rows.
// -------------------------------------------------------------------------
__launch_bounds__(192)
__global__ void corr_main(const float* __restrict__ x1,
                          const bf16_t* __restrict__ x2t,
                          float* __restrict__ out) {
  extern __shared__ __attribute__((aligned(16))) char smem[];
  bf16_t* Blds  = (bf16_t*)smem;                   // [2][96][BSTR] double buffer
  bf16_t* Alds  = (bf16_t*)(smem + ALDS_OFF);      // [32][BSTR]
  float*  Gband = (float*)(smem + GBAND_OFF);      // [2][KD][GSTR] + dump

  const int tid  = threadIdx.x;
  const int lane = tid & 31;
  const int wv   = tid >> 5;          // 0..5 : N-tile owned by this wave

  const int wg = blockIdx.x;          // 384 = 2 * 32 * 6
  const int wt = wg % 6;
  const int hp = (wg / 6) % 32;
  const int b  = wg / (6 * 32);
  const int w0 = wt * 16;
  const int h0 = hp * 2;

  const float scale = 0.08838834764831845f;   // 1/sqrt(128)

  // byte address of x2t[b][r=0][w0][0]
  const uint64_t x2t_base = (uint64_t)(uintptr_t)(x2t + (((size_t)b * H_) * XP + w0) * C_);
  const int rlo = (h0 - MD) > 0 ? (h0 - MD) : 0;
  const int rhi = (h0 + 1 + MD) < (H_ - 1) ? (h0 + 1 + MD) : (H_ - 1);

  // ---- kick off the first TDM immediately: overlaps prologue work ---------
  if (wv == 0) {
    tdm_load_bslice(x2t_base + (uint64_t)rlo * (XP * C_ * 2), 0u);
  }

  // ---- zero the output slabs whose x2 row is entirely padding --------------
  for (int hi = 0; hi < 2; ++hi) {
    const int h = h0 + hi;
    for (int dy = 0; dy < KD; ++dy) {
      const int r = h + dy - MD;
      if (r >= 0 && r < H_) continue;
      float4 z = make_float4(0.f, 0.f, 0.f, 0.f);
      for (int idx = tid; idx < KD * 4; idx += 192) {
        const int dx = idx >> 2, q = idx & 3;
        *(float4*)(out + ((((size_t)b * (KD * KD) + dy * KD + dx) * H_ + h) * W_ + w0 + q * 4)) = z;
      }
    }
  }

  // ---- A tile -> LDS: Alds[m][c], m = hi*16 + i  (h = h0+hi, w = w0+i) -----
  for (int idx = tid; idx < 32 * C_; idx += 192) {
    const int m = idx & 31;
    const int c = idx >> 5;
    const float v = x1[(((size_t)b * C_ + c) * H_ + h0 + (m >> 4)) * W_ + w0 + (m & 15)];
    Alds[m * BSTR + c] = (bf16_t)v;
  }
  __syncthreads();

  // ---- persistent A fragments (ISA 16-bit A layout: lanes0-15 K+0, 16-31 K+8)
  const int lm  = lane & 15;
  const int kkA = (lane >> 4) * 8;
  v16bf afrag[2][4];
#pragma unroll
  for (int mt = 0; mt < 2; ++mt)
#pragma unroll
    for (int kc = 0; kc < 4; ++kc) {
      const bf16_t* p = Alds + (mt * 16 + lm) * BSTR + kc * 32 + kkA;
      v8bf l8 = *(const v8bf*)p;           // local K = kk .. kk+7
      v8bf h8 = *(const v8bf*)(p + 16);    // local K = 16+kk .. 16+kk+7
      afrag[mt][kc] = __builtin_shufflevector(l8, h8, 0,1,2,3,4,5,6,7,8,9,10,11,12,13,14,15);
    }

  const int ln  = lane & 15;
  const int kkB = (lane >> 4) * 8;      // B layout: lanes0-15 K+0, lanes16-31 K+8
  const int ibase = (lane >> 4) << 3;   // C/D layout: row = v + 8*(lane>=16)
  const int j     = wv * 16 + ln;       // global column of this lane

  for (int r = rlo; r <= rhi; ++r) {
    const int cur = (r - rlo) & 1;
    __syncthreads();   // prev Gband consumed; buf[1-cur] reads (iter r-1) done

    if (wv == 0) {
      if (r < rhi) {
        // prefetch next row slice into the other buffer, then wait for current
        tdm_load_bslice(x2t_base + (uint64_t)(r + 1) * (XP * C_ * 2),
                        cur ? 0u : (unsigned)BBUF_BYTES);
        __builtin_amdgcn_s_wait_tensorcnt(1);
      } else {
        __builtin_amdgcn_s_wait_tensorcnt(0);
      }
    }
    __syncthreads();

    // --- 8 WMMAs: two 16x16 M-tiles x 4 k-chunks, this wave's 16-col N-tile
    v8f acc0 = {0.f, 0.f, 0.f, 0.f, 0.f, 0.f, 0.f, 0.f};
    v8f acc1 = {0.f, 0.f, 0.f, 0.f, 0.f, 0.f, 0.f, 0.f};
    const bf16_t* Bbuf = Blds + cur * BBUF_ELEMS;
#pragma unroll
    for (int kc = 0; kc < 4; ++kc) {
      const bf16_t* p = Bbuf + (wv * 16 + ln) * BSTR + kc * 32 + kkB;
      v8bf l8 = *(const v8bf*)p;
      v8bf h8 = *(const v8bf*)(p + 16);
      v16bf bfrag = __builtin_shufflevector(l8, h8, 0,1,2,3,4,5,6,7,8,9,10,11,12,13,14,15);
      acc0 = __builtin_amdgcn_wmma_f32_16x16x32_bf16(false, afrag[0][kc], false, bfrag,
                                                     (short)0, acc0, false, false);
      acc1 = __builtin_amdgcn_wmma_f32_16x16x32_bf16(false, afrag[1][kc], false, bfrag,
                                                     (short)0, acc1, false, false);
    }

    // --- branchless diagonal-band scatter: out-of-band lanes hit a dump slot
#pragma unroll
    for (int v = 0; v < 8; ++v) {
      const int i   = ibase + v;
      const int dx  = j - i;
      const bool ok = (unsigned)dx < (unsigned)KD;
      const int o0  = ok ? (dx * GSTR + i) : GDUMP;
      const int o1  = ok ? ((KD + dx) * GSTR + i) : GDUMP;
      Gband[o0] = acc0[v] * scale;
      Gband[o1] = acc1[v] * scale;
    }
    __syncthreads();

    // --- coalesced float4 stores of the two (h, dy) slabs ---------------------
    for (int hi = 0; hi < 2; ++hi) {
      const int dy = r + MD - (h0 + hi);
      if (dy < 0 || dy >= KD) continue;
      for (int idx = tid; idx < KD * 4; idx += 192) {
        const int dx = idx >> 2, q = idx & 3;
        float4 vv = *(const float4*)&Gband[(hi * KD + dx) * GSTR + q * 4];
        *(float4*)(out + ((((size_t)b * (KD * KD) + dy * KD + dx) * H_ + h0 + hi) * W_ + w0 + q * 4)) = vv;
      }
    }
  }
}

// -------------------------------------------------------------------------
// Scalar fallback (only if workspace is too small for the bf16 transpose).
// -------------------------------------------------------------------------
__global__ void corr_fallback(const float* __restrict__ x1, const float* __restrict__ x2,
                              float* __restrict__ out, int n) {
  int o = blockIdx.x * 256 + threadIdx.x;
  if (o >= n) return;
  int w = o % W_;
  int h = (o / W_) % H_;
  int s = (o / (W_ * H_)) % (KD * KD);
  int b = o / (W_ * H_ * KD * KD);
  int dy = s / KD, dx = s % KD;
  int r = h + dy - MD, col = w + dx - MD;
  float acc = 0.0f;
  if (r >= 0 && r < H_ && col >= 0 && col < W_) {
    for (int c = 0; c < C_; ++c)
      acc += x1[(((size_t)b * C_ + c) * H_ + h) * W_ + w] *
             x2[(((size_t)b * C_ + c) * H_ + r) * W_ + col];
  }
  out[o] = acc * 0.08838834764831845f;
}

extern "C" void kernel_launch(void* const* d_in, const int* in_sizes, int n_in,
                              void* d_out, int out_size, void* d_ws, size_t ws_size,
                              hipStream_t stream) {
  const float* x1 = (const float*)d_in[0];
  const float* x2 = (const float*)d_in[1];
  float* out = (float*)d_out;

  const size_t ws_needed = (size_t)B_ * H_ * XP * C_ * sizeof(bf16_t);  // 5,767,168 B
  if (ws_size < ws_needed) {
    corr_fallback<<<(out_size + 255) / 256, 256, 0, stream>>>(x1, x2, out, out_size);
    return;
  }

  bf16_t* x2t = (bf16_t*)d_ws;
  corr_prepass<<<(B_ * H_ * XP * C_) / 256, 256, 0, stream>>>(x2, x2t);

  corr_main<<<B_ * 32 * 6, 192, LDS_TOTAL, stream>>>(x1, x2t, out);
}
